// LOGM_63900523430697
// MI455X (gfx1250) — compile-verified, matrix operand order checked by reading, not run on hardware
//
#include <hip/hip_runtime.h>
#include <hip/hip_bf16.h>

// ---------------------------------------------------------------------------
// Types for CDNA5 WMMA (wave32, 16x16x32 bf16 -> f32 accumulate)
// ---------------------------------------------------------------------------
typedef __attribute__((ext_vector_type(16))) __bf16 v16bf;
typedef __attribute__((ext_vector_type(8)))  float  v8f;

union BF16x16 { unsigned int u[8]; uint4 q[2]; v16bf v; };

// zigzag tables from the reference
__constant__ int cLOW[64] = {
  0,1,8,16,9,2,3,10,17,24,32,25,18,11,4,5,12,19,26,33,40,48,41,34,27,20,13,6,
  7,14,21,28,35,42,49,56,57,50,43,36,29,22,15,23,30,37,44,51,58,59,52,45,38,31,
  39,46,53,60,61,54,47,55,62,63};
__constant__ int cREV[64] = {
  0,1,5,6,14,15,27,28,2,4,7,13,16,26,29,42,3,8,12,17,25,30,41,43,9,11,18,24,
  31,40,44,53,10,19,23,32,39,45,52,54,20,22,33,38,46,51,55,60,21,34,37,47,50,
  56,59,61,35,36,48,49,57,58,62,63};

__device__ __forceinline__ unsigned short f2bf(float x) {
  unsigned int u = __float_as_uint(x);
  unsigned int r = (u + 0x7FFFu + ((u >> 16) & 1u)) >> 16;   // RNE
  return (unsigned short)r;
}
__device__ __forceinline__ float bf2f(unsigned short h) {
  return __uint_as_float(((unsigned int)h) << 16);
}

// ---------------------------------------------------------------------------
// 1) FiLM MLP: qt_flat(16x128) -> 512 -> 512 -> 512 -> gamma/beta (16x320)
// ---------------------------------------------------------------------------
__global__ void film_mlp(const float* __restrict__ QT,
                         const float* __restrict__ W1, const float* __restrict__ b1,
                         const float* __restrict__ W2, const float* __restrict__ b2,
                         const float* __restrict__ W3, const float* __restrict__ b3,
                         const float* __restrict__ Wg, const float* __restrict__ bg,
                         const float* __restrict__ Wb, const float* __restrict__ bb,
                         float* __restrict__ gamma, float* __restrict__ beta) {
  __shared__ float sa[512];
  __shared__ float sb[512];
  const int t = threadIdx.x;   // 512 threads
  for (int b = 0; b < 16; ++b) {
    if (t < 128) sa[t] = QT[b * 192 + t];     // first two 8x8 tables
    __syncthreads();
    float acc = b1[t];
    for (int k = 0; k < 128; ++k) acc = fmaf(sa[k], W1[k * 512 + t], acc);
    sb[t] = fmaxf(acc, 0.f);
    __syncthreads();
    acc = b2[t];
    for (int k = 0; k < 512; ++k) acc = fmaf(sb[k], W2[k * 512 + t], acc);
    sa[t] = fmaxf(acc, 0.f);
    __syncthreads();
    acc = b3[t];
    for (int k = 0; k < 512; ++k) acc = fmaf(sa[k], W3[k * 512 + t], acc);
    sb[t] = fmaxf(acc, 0.f);
    __syncthreads();
    if (t < 320) {
      float g = bg[t], bt = bb[t];
      for (int k = 0; k < 512; ++k) {
        float e = sb[k];
        g  = fmaf(e, Wg[k * 320 + t], g);
        bt = fmaf(e, Wb[k * 320 + t], bt);
      }
      gamma[b * 320 + t] = 1.f / (1.f + expf(-g));
      beta[b * 320 + t]  = tanhf(bt);
    }
    __syncthreads();
  }
}

// ---------------------------------------------------------------------------
// 2a) W^T to bf16, row-major [o][c] (K-pairs contiguous)
// ---------------------------------------------------------------------------
__global__ void prep_w(const float* __restrict__ Wconv, unsigned short* __restrict__ WT) {
  int idx = blockIdx.x * 256 + threadIdx.x;    // 320*320
  int o = idx / 320, c = idx - o * 320;
  WT[o * 320 + c] = f2bf(Wconv[c * 320 + o]);
}

// ---------------------------------------------------------------------------
// 2b) Build Dl^T, bf16, K-pair interleaved: u32[(c>>1)*65536 + n] = {c, c+1}
//     n = b*4096 + i*64 + j,  Dl[b,c,i,j] per reference zigzag/dequant rules.
// ---------------------------------------------------------------------------
__global__ void pack_dl(const float* __restrict__ Y, const float* __restrict__ C,
                        const float* __restrict__ QT, unsigned int* __restrict__ Dl2) {
  unsigned int idx = blockIdx.x * 256 + threadIdx.x;   // z*65536 + n, z<160
  int n = idx & 65535;
  int z = idx >> 16;
  int b = n >> 12;
  int p = n & 4095;
  int i = p >> 6, j = p & 63;
  unsigned int out = 0;
#pragma unroll
  for (int hh = 0; hh < 2; ++hh) {
    int c = z * 2 + hh;
    float val;
    if (c < 224) {                 // luma: 4 planes x 56 low freqs
      int q = c / 56, l = c - q * 56;
      int f = cLOW[l];
      int a1 = q >> 1, a2 = q & 1;
      int yi = ((b * 128 + (i * 2 + a1)) * 128 + (j * 2 + a2)) * 64 + f;
      val = Y[yi] * QT[b * 192 + f];
    } else {                       // chroma: 2 planes x 48 low freqs
      int cc = c - 224;
      int u = cc / 48, l = cc - u * 48;
      int f = cLOW[l];
      int ci = (((b * 2 + u) * 64 + i) * 64 + j) * 64 + f;
      val = C[ci] * QT[b * 192 + 64 + u * 64 + f];
    }
    out |= ((unsigned int)f2bf(val)) << (16 * hh);
  }
  Dl2[idx] = out;   // pair u32 index == z*65536+n == idx
}

// ---------------------------------------------------------------------------
// 3) GEMM + FiLM:  G[m,n] = gamma[b,m]*(sum_c WT[m,c]*Dl[c,n] + bconv[m]) + beta[b,m]
//    M=320 (20 tiles), K=320 (10 steps of 32), N=65536 (one 16-wide tile/wave).
//    Each wave stages its 320x16 Dl slab in LDS (transposed, padded stride 164
//    dwords -> conflict-free banks, 16B-aligned rows) and reads two
//    ds_load_b128 per WMMA.  No VGPR spills; A rows stream from L2.
// ---------------------------------------------------------------------------
#define SLAB_STRIDE 164                    // dwords per n-row (160 used + 4 pad)
#define SLAB_DW     (16 * SLAB_STRIDE)     // dwords per wave slab

__global__ __launch_bounds__(256) void gemm_film(
    const unsigned int* __restrict__ Dl2,
    const unsigned int* __restrict__ WT2,
    const float* __restrict__ bconv,
    const float* __restrict__ gamma,
    const float* __restrict__ beta,
    unsigned short* __restrict__ G) {
  __shared__ unsigned int sB[8 * SLAB_DW];          // 8 waves * 10.25KB = 82KB
  const int lane  = threadIdx.x & 31;
  const int wave  = threadIdx.x >> 5;
  const int ntile = blockIdx.x * 8 + wave;          // 0..4095
  const int n0    = ntile * 16;
  const int b     = n0 >> 12;                       // whole tile in one batch
  const int half  = lane >> 4;
  const int r     = lane & 15;
  unsigned int* slab = sB + wave * SLAB_DW;

  // ---- Fill this wave's slab: LDS[n][pair] = Dl2[pair*65536 + n0+n] --------
  {
    const int prow0 = lane >> 2;                    // 0..7
    const int nq    = (lane & 3) * 4;               // 0,4,8,12
    for (int t = 0; t < 20; ++t) {
      int p = t * 8 + prow0;                        // pair row 0..159
      const uint4 d = *(const uint4*)(Dl2 + (unsigned int)p * 65536u +
                                      (unsigned int)(n0 + nq));
      slab[(nq + 0) * SLAB_STRIDE + p] = d.x;
      slab[(nq + 1) * SLAB_STRIDE + p] = d.y;
      slab[(nq + 2) * SLAB_STRIDE + p] = d.z;
      slab[(nq + 3) * SLAB_STRIDE + p] = d.w;
    }
  }
  __syncthreads();

  // ---- 20 M-tiles x 10 K-steps of v_wmma_f32_16x16x32_bf16 -----------------
  for (int mt = 0; mt < 20; ++mt) {
    const int m0 = mt * 16;
    if (mt + 1 < 20)  // gfx1250 global_prefetch_b8 for next A row block
      __builtin_prefetch(WT2 + (unsigned int)((mt + 1) * 16 + r) * 160u, 0, 1);
    v8f acc = {0.f, 0.f, 0.f, 0.f, 0.f, 0.f, 0.f, 0.f};
#pragma unroll
    for (int ks = 0; ks < 10; ++ks) {
      // A fragment: row m0+r of W^T, K-pairs ks*16 + 4*half + {0..3, 8..11}
      BF16x16 af;
      const unsigned int* asrc =
          WT2 + (unsigned int)(m0 + r) * 160u + ks * 16 + 4 * half;
      af.q[0] = *(const uint4*)asrc;
      af.q[1] = *(const uint4*)(asrc + 8);
      // B fragment: col n0+r of Dl, same K-pairs, from LDS (2x ds_load_b128)
      BF16x16 bf_;
      const unsigned int* bsrc = slab + r * SLAB_STRIDE + ks * 16 + 4 * half;
      bf_.q[0] = *(const uint4*)bsrc;
      bf_.q[1] = *(const uint4*)(bsrc + 8);
      acc = __builtin_amdgcn_wmma_f32_16x16x32_bf16(
          false, af.v, false, bf_.v, (short)0, acc, false, false);
    }
    // fused bias + FiLM, store bf16.  C/D layout: elem v -> M = v + 8*half, N = r.
#pragma unroll
    for (int v = 0; v < 8; ++v) {
      int m = m0 + v + 8 * half;
      float val = gamma[b * 320 + m] * (acc[v] + bconv[m]) + beta[b * 320 + m];
      G[(unsigned int)m * 65536u + (unsigned int)(n0 + r)] = f2bf(val);
    }
  }
}

// ---------------------------------------------------------------------------
// 4) Epilogues: inverse zigzag scatter, 0.5*tanh + residual, output transforms
// ---------------------------------------------------------------------------
__global__ void epi_y(const float* __restrict__ Y, const float* __restrict__ QT,
                      const unsigned short* __restrict__ G,
                      float* __restrict__ Yout, float* __restrict__ Ypred) {
  unsigned int idx = blockIdx.x * 256 + threadIdx.x;   // 16,777,216
  int f  = idx & 63;
  int x8 = (idx >> 6) & 127;
  int y8 = (idx >> 13) & 127;
  int b  = idx >> 20;
  float yraw = Y[idx];
  int l = cREV[f];
  float yp = yraw;
  if (l < 56) {
    int q = (y8 & 1) * 2 + (x8 & 1);
    int m = q * 56 + l;
    int n = (b << 12) + ((y8 >> 1) << 6) + (x8 >> 1);
    yp = 0.5f * tanhf(bf2f(G[(unsigned int)m * 65536u + (unsigned int)n])) + yraw;
  }
  Ypred[idx] = yp;
  Yout[idx]  = yp * QT[b * 192 + f];
}

__global__ void epi_c(const float* __restrict__ C, const float* __restrict__ QT,
                      const unsigned short* __restrict__ G,
                      float* __restrict__ Cout, float* __restrict__ Cpred) {
  unsigned int idx = blockIdx.x * 256 + threadIdx.x;   // 8,388,608
  int f = idx & 63;
  int j = (idx >> 6) & 63;
  int i = (idx >> 12) & 63;
  int u = (idx >> 18) & 1;
  int b = idx >> 19;
  float craw = C[idx];
  int l = cREV[f];
  float cp = craw;
  if (l < 48) {
    int m = 224 + u * 48 + l;
    int n = (b << 12) + (i << 6) + j;
    cp = 0.5f * tanhf(bf2f(G[(unsigned int)m * 65536u + (unsigned int)n])) + craw;
  }
  Cpred[idx] = cp;
  Cout[idx]  = cp * QT[b * 192 + 64 + (u << 6) + f];
}

// ---------------------------------------------------------------------------
// Launch
// ---------------------------------------------------------------------------
extern "C" void kernel_launch(void* const* d_in, const int* in_sizes, int n_in,
                              void* d_out, int out_size, void* d_ws, size_t ws_size,
                              hipStream_t stream) {
  const float* Y     = (const float*)d_in[0];
  const float* C     = (const float*)d_in[1];
  const float* QT    = (const float*)d_in[2];
  const float* W1    = (const float*)d_in[3];
  const float* b1    = (const float*)d_in[4];
  const float* W2    = (const float*)d_in[5];
  const float* b2    = (const float*)d_in[6];
  const float* W3    = (const float*)d_in[7];
  const float* b3    = (const float*)d_in[8];
  const float* Wg    = (const float*)d_in[9];
  const float* bg    = (const float*)d_in[10];
  const float* Wb    = (const float*)d_in[11];
  const float* bb    = (const float*)d_in[12];
  const float* Wconv = (const float*)d_in[13];
  const float* bconv = (const float*)d_in[14];

  char* ws = (char*)d_ws;
  float*          gamma = (float*)(ws + 0);                // 16*320*4  = 20480
  float*          beta  = (float*)(ws + 20480);            // 20480
  unsigned int*   Dl2   = (unsigned int*)(ws + 40960);     // 320*65536*2 = 41,943,040
  unsigned int*   WT2   = (unsigned int*)(ws + 41984000);  // 320*320*2   = 204,800
  unsigned short* G     = (unsigned short*)(ws + 42188800);// 320*65536*2 = 41,943,040

  float* Yout  = (float*)d_out;            // 16,777,216
  float* Cout  = Yout + 16777216;          //  8,388,608
  float* Ypred = Cout + 8388608;           // 16,777,216
  float* Cpred = Ypred + 16777216;         //  8,388,608

  film_mlp<<<1, 512, 0, stream>>>(QT, W1, b1, W2, b2, W3, b3, Wg, bg, Wb, bb, gamma, beta);
  prep_w  <<<400, 256, 0, stream>>>(Wconv, (unsigned short*)WT2);
  pack_dl <<<40960, 256, 0, stream>>>(Y, C, QT, Dl2);
  gemm_film<<<512, 256, 0, stream>>>(Dl2, WT2, bconv, gamma, beta, G);
  epi_y   <<<65536, 256, 0, stream>>>(Y, QT, G, Yout, Ypred);
  epi_c   <<<32768, 256, 0, stream>>>(C, QT, G, Cout, Cpred);
}
